// CopyLSTMDecoder_8392366096742
// MI455X (gfx1250) — matrix operand order repaired and checked by reading, not compiled, not run on hardware
//
#include <hip/hip_runtime.h>
#include <hip/hip_bf16.h>
#include <stdint.h>

// Problem constants (match reference)
#define VSZ   32000
#define VEXT  32100
#define EMB   256
#define HID   256
#define DENC  512
#define BB    32
#define LL    512
#define TT    64

typedef __attribute__((ext_vector_type(16))) __bf16 bf16x16;
typedef __attribute__((ext_vector_type(8)))  float  f32x8;

union ABf { bf16x16 v; uint4 q[2]; };

// A-fragment (16x32 bf16, M x K): lane m (0..15) / m+16 : halves 0..7 = K kh*8+(0..7),
// halves 8..15 = K 16+kh*8+(0..7)   (kh = lane>>4)
__device__ __forceinline__ bf16x16 load_a_frag(const __hip_bfloat16* base, int row, int lda,
                                               int k0, int kh) {
  ABf f;
  const __hip_bfloat16* p = base + (size_t)row * lda + k0 + kh * 8;
  f.q[0] = *(const uint4*)(const void*)(p);
  f.q[1] = *(const uint4*)(const void*)(p + 16);
  return f.v;
}

// B-fragment (32x16 bf16, K x N): lane holds column (lane&15); halves 0..7 = K kh*8+(0..7), halves 8..15 = K 16+kh*8+(0..7).
// wrow points at the weight row for this output column (K-contiguous).
__device__ __forceinline__ bf16x16 load_b_frag(const __hip_bfloat16* wrow, int k0, int kh) {
  ABf f;
  const __hip_bfloat16* p = wrow + k0 + kh * 8;
  f.q[0] = *(const uint4*)(const void*)(p);
  f.q[1] = *(const uint4*)(const void*)(p + 16);
  return f.v;
}

__device__ __forceinline__ f32x8 wmma_bf16(bf16x16 a, bf16x16 b, f32x8 c) {
  return __builtin_amdgcn_wmma_f32_16x16x32_bf16(false, a, false, b, (short)0, c, false, false);
}

__device__ __forceinline__ float sigm(float x) { return 1.0f / (1.0f + __expf(-x)); }

// order-preserving float<->uint for atomicMax on floats
__device__ __forceinline__ unsigned ford(float f) {
  unsigned u = __float_as_uint(f);
  return (u & 0x80000000u) ? ~u : (u | 0x80000000u);
}
__device__ __forceinline__ float fuord(unsigned u) {
  return __uint_as_float((u & 0x80000000u) ? (u & 0x7fffffffu) : ~u);
}

// ---------------------------------------------------------------------------
// K0: one-time (per call) prep: bf16 weight conversions, gate-concat layouts,
// fused biases, state initialization.  Deterministic, re-run every call.
// ---------------------------------------------------------------------------
__global__ __launch_bounds__(256) void prep_kernel(
    const float* __restrict__ emb, const float* __restrict__ Wih0,
    const float* __restrict__ Whh0, const float* __restrict__ bih0,
    const float* __restrict__ bhh0, const float* __restrict__ Wih1,
    const float* __restrict__ Whh1, const float* __restrict__ bih1,
    const float* __restrict__ bhh1, const float* __restrict__ projw,
    const float* __restrict__ h0, const float* __restrict__ c0,
    const float* __restrict__ pv0,
    __hip_bfloat16* embbf, __hip_bfloat16* wcat0, __hip_bfloat16* wcat1,
    __hip_bfloat16* projbf, float* bias0, float* bias1,
    float* h0s, float* h1s, float* c0s, float* c1s, float* pvout) {
  size_t stride = (size_t)gridDim.x * blockDim.x;
  size_t tid0 = (size_t)blockIdx.x * blockDim.x + threadIdx.x;
  for (size_t i = tid0; i < (size_t)VSZ * EMB; i += stride)
    embbf[i] = __float2bfloat16(emb[i]);
  for (size_t i = tid0; i < (size_t)1024 * 768; i += stride) {
    int n = (int)(i / 768), k = (int)(i % 768);
    float v = (k < 512) ? Wih0[n * 512 + k] : Whh0[n * 256 + (k - 512)];
    wcat0[i] = __float2bfloat16(v);
  }
  for (size_t i = tid0; i < (size_t)1024 * 512; i += stride) {
    int n = (int)(i / 512), k = (int)(i % 512);
    float v = (k < 256) ? Wih1[n * 256 + k] : Whh1[n * 256 + (k - 256)];
    wcat1[i] = __float2bfloat16(v);
  }
  for (size_t i = tid0; i < (size_t)256 * 768; i += stride)
    projbf[i] = __float2bfloat16(projw[i]);
  for (size_t i = tid0; i < 1024; i += stride) {
    bias0[i] = bih0[i] + bhh0[i];
    bias1[i] = bih1[i] + bhh1[i];
  }
  for (size_t i = tid0; i < (size_t)BB * HID; i += stride) {
    h0s[i] = h0[i];
    h1s[i] = h0[(size_t)BB * HID + i];
    c0s[i] = c0[i];
    c1s[i] = c0[(size_t)BB * HID + i];
    pvout[i] = pv0[i];
  }
}

// ---------------------------------------------------------------------------
// K1a: build bf16 A-operands for the LSTM GEMMs:
//   xcat0[b, 0:256]=emb(tok) | [256:512]=prev_out | [512:768]=h0
//   xcat1[b, 256:512]=h1   (cols 0:256 filled by layer-0 epilogue)
// ---------------------------------------------------------------------------
__global__ __launch_bounds__(256) void build_xcat_kernel(
    const float* __restrict__ embedding, const int* __restrict__ abstract,
    const float* __restrict__ pvout, const float* __restrict__ h0s,
    const float* __restrict__ h1s, __hip_bfloat16* xcat0,
    __hip_bfloat16* xcat1, int t) {
  int b = blockIdx.x, tid = threadIdx.x;
  int tok = abstract[b * TT + t];
  for (int k = tid; k < 768; k += 256) {
    float v = (k < 256) ? embedding[(size_t)tok * EMB + k]
            : (k < 512) ? pvout[b * EMB + (k - 256)]
                        : h0s[b * HID + (k - 512)];
    xcat0[b * 768 + k] = __float2bfloat16(v);
  }
  xcat1[b * 512 + 256 + tid] = __float2bfloat16(h1s[b * HID + tid]);
}

// ---------------------------------------------------------------------------
// K1/K2: WMMA LSTM cell. grid=4 blocks x 128 thr (4 waves); each wave owns 16
// hidden units -> 4 gate n-tiles x 2 m-tiles, fp32 accumulate, gate math per
// lane (i,f,g,o for (b,u) land in one lane via the D layout).
// ---------------------------------------------------------------------------
__global__ __launch_bounds__(128) void lstm_wmma_kernel(
    const __hip_bfloat16* __restrict__ A, int lda, int K,
    const __hip_bfloat16* __restrict__ W, const float* __restrict__ bias,
    float* cstate, float* hstate, __hip_bfloat16* outbf, int out_stride) {
  int wave = threadIdx.x >> 5, lane = threadIdx.x & 31;
  int ubase = blockIdx.x * 64 + wave * 16;
  int col = lane & 15, kh = lane >> 4;

  f32x8 acc[4][2] = {};
  for (int k0 = 0; k0 < K; k0 += 32) {
    bf16x16 a0 = load_a_frag(A, col, lda, k0, kh);
    bf16x16 a1 = load_a_frag(A, col + 16, lda, k0, kh);
#pragma unroll
    for (int g = 0; g < 4; ++g) {
      const __hip_bfloat16* wrow = W + (size_t)(g * 256 + ubase + col) * K;
      bf16x16 bb = load_b_frag(wrow, k0, kh);
      acc[g][0] = wmma_bf16(a0, bb, acc[g][0]);
      acc[g][1] = wmma_bf16(a1, bb, acc[g][1]);
    }
  }
  int u = ubase + col;
  float bi = bias[u], bf_ = bias[256 + u], bg = bias[512 + u], bo = bias[768 + u];
#pragma unroll
  for (int mt = 0; mt < 2; ++mt) {
#pragma unroll
    for (int r = 0; r < 8; ++r) {
      int b = mt * 16 + kh * 8 + r;
      float gi = acc[0][mt][r] + bi;
      float gf = acc[1][mt][r] + bf_;
      float gg = acc[2][mt][r] + bg;
      float go = acc[3][mt][r] + bo;
      float c = cstate[b * HID + u];
      float cn = sigm(gf) * c + sigm(gi) * tanhf(gg);
      float hn = sigm(go) * tanhf(cn);
      cstate[b * HID + u] = cn;
      hstate[b * HID + u] = hn;
      outbf[b * out_stride + u] = __float2bfloat16(hn);
    }
  }
}

// ---------------------------------------------------------------------------
// K4: per-batch attention block (grid=32 x 256 thr): query matvec, masked
// softmax over L=512, context, copy gate, zero+scatter of copy probabilities.
// enc_mem/enc_proj are L2-resident (33 MB each, 192 MB L2).
// ---------------------------------------------------------------------------
__global__ __launch_bounds__(256) void attn_kernel(
    const float* __restrict__ enc_mem, const float* __restrict__ enc_proj,
    const uint8_t* __restrict__ mask, const int* __restrict__ extend_art,
    const int* __restrict__ abstract, const float* __restrict__ embedding,
    const float* __restrict__ attn_w, const float* __restrict__ v_c,
    const float* __restrict__ v_s, const float* __restrict__ v_i,
    const float* __restrict__ copy_b, const float* __restrict__ h1s,
    __hip_bfloat16* cat5, float* gatebuf, float* addb, int t) {
  int b = blockIdx.x, tid = threadIdx.x;
  int wave = tid >> 5, lane = tid & 31;
  __shared__ float s_q[DENC];
  __shared__ float s_att[LL];
  __shared__ float s_l[HID];
  __shared__ float s_red[8];
  __shared__ float s_scalar;
  __shared__ int s_any;

  // zero this batch's row of the copy-scatter buffer
  for (int i = tid; i < VEXT; i += 256) addb[(size_t)b * VEXT + i] = 0.f;
  if (tid == 0) s_any = 0;
  s_l[tid] = h1s[b * HID + tid];
  __syncthreads();

  int mm = (mask[b * LL + tid] ? 1 : 0) | (mask[b * LL + tid + 256] ? 1 : 0);
  if (mm) atomicOr(&s_any, 1);

  // query = lstm_out @ attn_w  (coalesced over d across threads)
#pragma unroll 1
  for (int pass = 0; pass < 2; ++pass) {
    int d = tid + pass * 256;
    float q = 0.f;
    for (int k = 0; k < HID; ++k) q += s_l[k] * attn_w[k * DENC + d];
    s_q[d] = q;
  }
  __syncthreads();
  int allm = (s_any == 0);

  // scores: each wave owns 64 rows of enc_proj[b]; lane-split dot + shuffle reduce
  for (int li = 0; li < 64; ++li) {
    int l = wave * 64 + li;
    const float* row = enc_proj + ((size_t)b * LL + l) * DENC + lane * 16;
    float p = 0.f;
#pragma unroll
    for (int j = 0; j < 16; ++j) p += s_q[lane * 16 + j] * row[j];
    for (int o = 16; o > 0; o >>= 1) p += __shfl_xor(p, o, 32);
    if (lane == 0) {
      float sc = mask[b * LL + l] ? p : -1e18f;
      s_att[l] = allm ? 0.f : sc;
    }
  }
  __syncthreads();

  // softmax over 512
  float lm = fmaxf(s_att[tid], s_att[tid + 256]);
  for (int o = 16; o > 0; o >>= 1) lm = fmaxf(lm, __shfl_xor(lm, o, 32));
  if (lane == 0) s_red[wave] = lm;
  __syncthreads();
  if (tid == 0) {
    float m = s_red[0];
    for (int i = 1; i < 8; ++i) m = fmaxf(m, s_red[i]);
    s_scalar = m;
  }
  __syncthreads();
  float mx = s_scalar;
  float e0 = __expf(s_att[tid] - mx), e1 = __expf(s_att[tid + 256] - mx);
  s_att[tid] = e0;
  s_att[tid + 256] = e1;
  float ls = e0 + e1;
  for (int o = 16; o > 0; o >>= 1) ls += __shfl_xor(ls, o, 32);
  if (lane == 0) s_red[wave] = ls;
  __syncthreads();
  if (tid == 0) {
    float s = 0.f;
    for (int i = 0; i < 8; ++i) s += s_red[i];
    s_scalar = s;
  }
  __syncthreads();
  float inv = 1.f / s_scalar;
  s_att[tid] *= inv;
  s_att[tid + 256] *= inv;
  __syncthreads();

  // context + gate partials
  float gp = 0.f;
#pragma unroll 1
  for (int pass = 0; pass < 2; ++pass) {
    int d = tid + pass * 256;
    float c = 0.f;
    for (int l = 0; l < LL; ++l) c += s_att[l] * enc_mem[((size_t)b * LL + l) * DENC + d];
    cat5[b * 768 + 256 + d] = __float2bfloat16(c);
    gp += c * v_c[d];
  }
  int tok = abstract[b * TT + t];
  gp += s_l[tid] * v_s[tid] + embedding[(size_t)tok * EMB + tid] * v_i[tid];
  for (int o = 16; o > 0; o >>= 1) gp += __shfl_xor(gp, o, 32);
  if (lane == 0) s_red[wave] = gp;
  __syncthreads();
  if (tid == 0) {
    float s = 0.f;
    for (int i = 0; i < 8; ++i) s += s_red[i];
    float g = 1.f / (1.f + __expf(-(s + copy_b[0])));
    s_scalar = g;
    gatebuf[b] = g;
  }
  __syncthreads();
  float g = s_scalar;

  // scatter copy probabilities (single block owns this batch row)
#pragma unroll 1
  for (int pass = 0; pass < 2; ++pass) {
    int l = tid + pass * 256;
    int v = extend_art[b * LL + l];
    if (v > VEXT - 1) v = VEXT - 1;
    atomicAdd(&addb[(size_t)b * VEXT + v], s_att[l] * g);
  }
}

// ---------------------------------------------------------------------------
// K5: dec_out = [lstm_out | ctx] @ proj_w^T + proj_b  (WMMA, 1 block, 4 waves)
// ---------------------------------------------------------------------------
__global__ __launch_bounds__(128) void proj_wmma_kernel(
    const __hip_bfloat16* __restrict__ A, const __hip_bfloat16* __restrict__ W,
    const float* __restrict__ pb, float* pvout, __hip_bfloat16* decbf) {
  int wave = threadIdx.x >> 5, lane = threadIdx.x & 31;
  int col = lane & 15, kh = lane >> 4;
  f32x8 acc[4][2] = {};
  for (int k0 = 0; k0 < 768; k0 += 32) {
    bf16x16 a0 = load_a_frag(A, col, 768, k0, kh);
    bf16x16 a1 = load_a_frag(A, col + 16, 768, k0, kh);
#pragma unroll
    for (int nt = 0; nt < 4; ++nt) {
      int n = wave * 64 + nt * 16 + col;
      bf16x16 bb = load_b_frag(W + (size_t)n * 768, k0, kh);
      acc[nt][0] = wmma_bf16(a0, bb, acc[nt][0]);
      acc[nt][1] = wmma_bf16(a1, bb, acc[nt][1]);
    }
  }
#pragma unroll
  for (int nt = 0; nt < 4; ++nt) {
    int n = wave * 64 + nt * 16 + col;
    float bn = pb[n];
#pragma unroll
    for (int mt = 0; mt < 2; ++mt) {
#pragma unroll
      for (int r = 0; r < 8; ++r) {
        int b = mt * 16 + kh * 8 + r;
        float v = acc[nt][mt][r] + bn;
        pvout[b * EMB + n] = v;
        decbf[b * EMB + n] = __float2bfloat16(v);
      }
    }
  }
}

// ---------------------------------------------------------------------------
// K6a: logits = dec_out @ embedding^T (the dominant GEMM, WMMA bf16) plus
// per-block softmax partials (max + sum-exp) per batch row.
// grid = 250 blocks x 128 thr; block covers 128 vocab columns.
// ---------------------------------------------------------------------------
__global__ __launch_bounds__(128) void logits_kernel(
    const __hip_bfloat16* __restrict__ decbf,
    const __hip_bfloat16* __restrict__ embbf, float* logits, float* pmax,
    float* psum) {
  __shared__ unsigned s_m[BB];
  __shared__ float s_s[BB];
  int wave = threadIdx.x >> 5, lane = threadIdx.x & 31;
  int col = lane & 15, kh = lane >> 4;
  int vb = blockIdx.x * 128 + wave * 32;
  if (threadIdx.x < BB) { s_m[threadIdx.x] = 0u; s_s[threadIdx.x] = 0.f; }
  __syncthreads();

  f32x8 acc[2][2] = {};
  for (int k0 = 0; k0 < EMB; k0 += 32) {
    bf16x16 a0 = load_a_frag(decbf, col, EMB, k0, kh);
    bf16x16 a1 = load_a_frag(decbf, col + 16, EMB, k0, kh);
#pragma unroll
    for (int nt = 0; nt < 2; ++nt) {
      int v = vb + nt * 16 + col;
      bf16x16 bb = load_b_frag(embbf + (size_t)v * EMB, k0, kh);
      acc[nt][0] = wmma_bf16(a0, bb, acc[nt][0]);
      acc[nt][1] = wmma_bf16(a1, bb, acc[nt][1]);
    }
  }
#pragma unroll
  for (int mt = 0; mt < 2; ++mt) {
#pragma unroll
    for (int r = 0; r < 8; ++r) {
      int b = mt * 16 + kh * 8 + r;
      float x0 = acc[0][mt][r], x1 = acc[1][mt][r];
      logits[(size_t)b * VSZ + vb + col] = x0;
      logits[(size_t)b * VSZ + vb + 16 + col] = x1;
      atomicMax(&s_m[b], ford(fmaxf(x0, x1)));
    }
  }
  __syncthreads();
#pragma unroll
  for (int mt = 0; mt < 2; ++mt) {
#pragma unroll
    for (int r = 0; r < 8; ++r) {
      int b = mt * 16 + kh * 8 + r;
      float m = fuord(s_m[b]);
      atomicAdd(&s_s[b], __expf(acc[0][mt][r] - m) + __expf(acc[1][mt][r] - m));
    }
  }
  __syncthreads();
  if (threadIdx.x < BB) {
    pmax[threadIdx.x * 256 + blockIdx.x] = fuord(s_m[threadIdx.x]);
    psum[threadIdx.x * 256 + blockIdx.x] = s_s[threadIdx.x];
  }
}

// K6b: merge per-block (m,s) partials -> global (M,S) per batch
__global__ void softmax_reduce_kernel(const float* __restrict__ pmax,
                                      const float* __restrict__ psum,
                                      float* mfin, float* sfin, int nblocks) {
  int b = threadIdx.x;
  if (b >= BB) return;
  float m = -3.4e38f;
  for (int i = 0; i < nblocks; ++i) m = fmaxf(m, pmax[b * 256 + i]);
  float s = 0.f;
  for (int i = 0; i < nblocks; ++i) s += psum[b * 256 + i] * __expf(pmax[b * 256 + i] - m);
  mfin[b] = m;
  sfin[b] = s;
}

// K6c: logp = log((1-gate)*softmax(logits) + copy_add + 1e-12); pad cols (v>=V)
// contribute exp(-1e8)/Z == 0 in fp32, so only the copy term survives there.
__global__ __launch_bounds__(256) void final_kernel(
    const float* __restrict__ logits, const float* __restrict__ addb,
    const float* __restrict__ gatebuf, const float* __restrict__ mfin,
    const float* __restrict__ sfin, float* out, int t) {
  unsigned idx = blockIdx.x * 256u + threadIdx.x;
  if (idx >= (unsigned)BB * VEXT) return;
  int b = idx / VEXT;
  int v = idx - b * VEXT;
  float a = addb[idx] + 1e-12f;
  if (v < VSZ)
    a += (1.f - gatebuf[b]) * __expf(logits[(size_t)b * VSZ + v] - mfin[b]) / sfin[b];
  out[((size_t)b * TT + t) * VEXT + v] = logf(a);
}

// ---------------------------------------------------------------------------
extern "C" void kernel_launch(void* const* d_in, const int* in_sizes, int n_in,
                              void* d_out, int out_size, void* d_ws,
                              size_t ws_size, hipStream_t stream) {
  (void)in_sizes; (void)n_in; (void)out_size; (void)ws_size;
  const float* enc_mem    = (const float*)d_in[0];
  const float* enc_proj   = (const float*)d_in[1];
  const uint8_t* mask     = (const uint8_t*)d_in[2];
  const int* extend_art   = (const int*)d_in[3];
  const float* h0         = (const float*)d_in[4];
  const float* c0         = (const float*)d_in[5];
  const float* prev_out0  = (const float*)d_in[6];
  const int* abstract     = (const int*)d_in[7];
  /* d_in[8] = extend_vsize scalar; static VEXT used */
  const float* embedding  = (const float*)d_in[9];
  const float* W_ih0 = (const float*)d_in[10];
  const float* W_hh0 = (const float*)d_in[11];
  const float* b_ih0 = (const float*)d_in[12];
  const float* b_hh0 = (const float*)d_in[13];
  const float* W_ih1 = (const float*)d_in[14];
  const float* W_hh1 = (const float*)d_in[15];
  const float* b_ih1 = (const float*)d_in[16];
  const float* b_hh1 = (const float*)d_in[17];
  const float* attn_w = (const float*)d_in[18];
  const float* proj_w = (const float*)d_in[19];
  const float* proj_b = (const float*)d_in[20];
  const float* v_c = (const float*)d_in[21];
  const float* v_s = (const float*)d_in[22];
  const float* v_i = (const float*)d_in[23];
  const float* copy_b = (const float*)d_in[24];
  float* out = (float*)d_out;

  char* ws = (char*)d_ws;
  size_t o = 0;
  auto take = [&](size_t bytes) -> char* {
    char* p = ws + o;
    o = (o + bytes + 511) & ~(size_t)511;
    return p;
  };
  __hip_bfloat16* embbf  = (__hip_bfloat16*)take((size_t)VSZ * EMB * 2);
  __hip_bfloat16* wcat0  = (__hip_bfloat16*)take((size_t)1024 * 768 * 2);
  __hip_bfloat16* wcat1  = (__hip_bfloat16*)take((size_t)1024 * 512 * 2);
  __hip_bfloat16* projbf = (__hip_bfloat16*)take((size_t)256 * 768 * 2);
  float* bias0 = (float*)take(1024 * 4);
  float* bias1 = (float*)take(1024 * 4);
  float* h0s   = (float*)take((size_t)BB * HID * 4);
  float* h1s   = (float*)take((size_t)BB * HID * 4);
  float* c0s   = (float*)take((size_t)BB * HID * 4);
  float* c1s   = (float*)take((size_t)BB * HID * 4);
  float* pvout = (float*)take((size_t)BB * EMB * 4);
  __hip_bfloat16* xcat0 = (__hip_bfloat16*)take((size_t)BB * 768 * 2);
  __hip_bfloat16* xcat1 = (__hip_bfloat16*)take((size_t)BB * 512 * 2);
  __hip_bfloat16* cat5  = (__hip_bfloat16*)take((size_t)BB * 768 * 2);
  __hip_bfloat16* decbf = (__hip_bfloat16*)take((size_t)BB * EMB * 2);
  float* gatebuf = (float*)take(BB * 4);
  float* logits  = (float*)take((size_t)BB * VSZ * 4);
  float* addb    = (float*)take((size_t)BB * VEXT * 4);
  float* pmax    = (float*)take((size_t)BB * 256 * 4);
  float* psum    = (float*)take((size_t)BB * 256 * 4);
  float* mfin    = (float*)take(BB * 4);
  float* sfin    = (float*)take(BB * 4);

  prep_kernel<<<1024, 256, 0, stream>>>(
      embedding, W_ih0, W_hh0, b_ih0, b_hh0, W_ih1, W_hh1, b_ih1, b_hh1,
      proj_w, h0, c0, prev_out0, embbf, wcat0, wcat1, projbf, bias0, bias1,
      h0s, h1s, c0s, c1s, pvout);

  const int NB6 = VSZ / 128;  // 250
  for (int t = 0; t < TT; ++t) {
    build_xcat_kernel<<<BB, 256, 0, stream>>>(embedding, abstract, pvout, h0s,
                                              h1s, xcat0, xcat1, t);
    lstm_wmma_kernel<<<4, 128, 0, stream>>>(xcat0, 768, 768, wcat0, bias0,
                                            c0s, h0s, xcat1, 512);
    lstm_wmma_kernel<<<4, 128, 0, stream>>>(xcat1, 512, 512, wcat1, bias1,
                                            c1s, h1s, cat5, 768);
    attn_kernel<<<BB, 256, 0, stream>>>(enc_mem, enc_proj, mask, extend_art,
                                        abstract, embedding, attn_w, v_c, v_s,
                                        v_i, copy_b, h1s, cat5, gatebuf, addb, t);
    proj_wmma_kernel<<<1, 128, 0, stream>>>(cat5, projbf, proj_b, pvout, decbf);
    logits_kernel<<<NB6, 128, 0, stream>>>(decbf, embbf, logits, pmax, psum);
    softmax_reduce_kernel<<<1, 32, 0, stream>>>(pmax, psum, mfin, sfin, NB6);
    final_kernel<<<(BB * VEXT + 255) / 256, 256, 0, stream>>>(
        logits, addb, gatebuf, mfin, sfin, out, t);
  }
}
